// Attention_21595095564898
// MI455X (gfx1250) — matrix operand order repaired and checked by reading, not compiled, not run on hardware
//
#include <hip/hip_runtime.h>

// ---------------------------------------------------------------------------
// Types for CDNA5 WMMA (wave32): v_wmma_f32_16x16x32_bf16
// ---------------------------------------------------------------------------
typedef __attribute__((ext_vector_type(16))) __bf16 bf16x16;
typedef __attribute__((ext_vector_type(8)))  __bf16 bf16x8;
typedef __attribute__((ext_vector_type(8)))  float  f32x8;

#define WMMA_BF16(a, b, c) \
    __builtin_amdgcn_wmma_f32_16x16x32_bf16(false, (a), false, (b), (short)0, (c), false, false)

static __device__ __forceinline__ bf16x16 frag_join(bf16x8 lo, bf16x8 hi) {
    bf16x16 v;
#pragma unroll
    for (int i = 0; i < 8; ++i) { v[i] = lo[i]; v[i + 8] = hi[i]; }
    return v;
}

// A-matrix fragment (16x32, MxK). Lane half h holds K = {k0+8h..} U {k0+16+8h..}.
static __device__ __forceinline__ bf16x16 load_frag_a(const __bf16* row, int k0, int half) {
    bf16x8 lo = *reinterpret_cast<const bf16x8*>(row + k0 + half * 8);
    bf16x8 hi = *reinterpret_cast<const bf16x8*>(row + k0 + 16 + half * 8);
    return frag_join(lo, hi);
}

// B-matrix fragment (32x16, KxN). Lane half h holds K = {k0+8h..} U {k0+16+8h..}.
static __device__ __forceinline__ bf16x16 load_frag_b(const __bf16* row, int k0, int half) {
    bf16x8 lo = *reinterpret_cast<const bf16x8*>(row + k0 + half * 8);
    bf16x8 hi = *reinterpret_cast<const bf16x8*>(row + k0 + 16 + half * 8);
    return frag_join(lo, hi);
}

// ---------------------------------------------------------------------------
// Async global -> LDS copy of one 16-byte chunk (falls back to reg bounce).
// Builtin signature (from hipcc diagnostic): param0 = v4i in AS(1) (__device__),
// param1 = v4i in AS(3) (__shared__), then i32 offset, i32 cpol.
// ---------------------------------------------------------------------------
#if defined(__has_builtin)
#if __has_builtin(__builtin_amdgcn_global_load_async_to_lds_b128)
#define HAS_ASYNC_LDS 1
#endif
#endif

typedef int v4i_vs __attribute__((vector_size(4 * sizeof(int))));
typedef __attribute__((address_space(1))) v4i_vs* as1_v4i_p;
typedef __attribute__((address_space(3))) v4i_vs* as3_v4i_p;

static __device__ __forceinline__ void copy16_to_lds(const __bf16* g, __bf16* l) {
#ifdef HAS_ASYNC_LDS
    __builtin_amdgcn_global_load_async_to_lds_b128((as1_v4i_p)(g), (as3_v4i_p)(l), 0, 0);
#else
    *reinterpret_cast<uint4*>(l) = *reinterpret_cast<const uint4*>(g);
#endif
}

static __device__ __forceinline__ void wait_async_copies() {
#ifdef HAS_ASYNC_LDS
#if __has_builtin(__builtin_amdgcn_s_wait_asynccnt)
    __builtin_amdgcn_s_wait_asynccnt(0);
#else
    asm volatile("s_wait_asynccnt 0" ::: "memory");
#endif
#endif
}

// ---------------------------------------------------------------------------
// fp32 -> bf16 conversion (4 elems / thread)
// ---------------------------------------------------------------------------
__global__ void f32_to_bf16_kernel(const float* __restrict__ in,
                                   __bf16* __restrict__ out, size_t n4) {
    size_t i = (size_t)blockIdx.x * blockDim.x + threadIdx.x;
    if (i >= n4) return;
    float4 v = reinterpret_cast<const float4*>(in)[i];
    __bf16 o[4] = { (__bf16)v.x, (__bf16)v.y, (__bf16)v.z, (__bf16)v.w };
    *reinterpret_cast<ulong1*>(out + 4 * i) = *reinterpret_cast<const ulong1*>(o);
}

// ---------------------------------------------------------------------------
// GEMM: C[m,n] = sum_k A[m,k] * B[n,k] (+ bias[n]).  A: [M,K] bf16, B: [N,K] bf16.
// Block = 256 threads = 8 waves; each wave owns a 32x32 tile (2x2 WMMA accs).
// ---------------------------------------------------------------------------
__global__ __launch_bounds__(256) void gemm_bf16_nt_kernel(
    const __bf16* __restrict__ A, const __bf16* __restrict__ B,
    const float* __restrict__ bias, float* __restrict__ C,
    int M, int N, int K) {
    const int lane = threadIdx.x & 31;
    const int wave = threadIdx.x >> 5;
    const int half = lane >> 4;
    const int r    = lane & 15;
    const int wm = wave & 3;
    const int wn = wave >> 2;
    const int m0 = blockIdx.y * 128 + wm * 32;
    const int n0 = blockIdx.x * 64  + wn * 32;

    const __bf16* arow0 = A + (size_t)(m0 + r)      * K;
    const __bf16* arow1 = A + (size_t)(m0 + 16 + r) * K;
    const __bf16* brow0 = B + (size_t)(n0 + r)      * K;
    const __bf16* brow1 = B + (size_t)(n0 + 16 + r) * K;

    f32x8 acc[2][2] = {};

    for (int k0 = 0; k0 < K; k0 += 32) {
        if (k0 + 64 < K) {
            __builtin_prefetch(arow0 + k0 + 64, 0, 3);
            __builtin_prefetch(brow0 + k0 + 64, 0, 3);
        }
        bf16x16 a0 = load_frag_a(arow0, k0, half);
        bf16x16 a1 = load_frag_a(arow1, k0, half);
        bf16x16 b0 = load_frag_b(brow0, k0, half);
        bf16x16 b1 = load_frag_b(brow1, k0, half);
        acc[0][0] = WMMA_BF16(a0, b0, acc[0][0]);
        acc[0][1] = WMMA_BF16(a0, b1, acc[0][1]);
        acc[1][0] = WMMA_BF16(a1, b0, acc[1][0]);
        acc[1][1] = WMMA_BF16(a1, b1, acc[1][1]);
    }

#pragma unroll
    for (int mt = 0; mt < 2; ++mt)
#pragma unroll
        for (int nt = 0; nt < 2; ++nt) {
            const int n = n0 + nt * 16 + r;
            const float bv = bias ? bias[n] : 0.0f;
#pragma unroll
            for (int j = 0; j < 8; ++j) {
                const int m = m0 + mt * 16 + j + 8 * half;
                C[(size_t)m * N + n] = acc[mt][nt][j] + bv;
            }
        }
}

// ---------------------------------------------------------------------------
// RoPE (full-embedding, adjacent pairs) on q,k + repack to bf16:
//   Qb, Kb: [B,H,T,D]   VTb: [B,H,D,T]
// ---------------------------------------------------------------------------
__global__ void rope_pack_kernel(const float* __restrict__ qf,
                                 const float* __restrict__ kf,
                                 const float* __restrict__ vf,
                                 __bf16* __restrict__ Qb,
                                 __bf16* __restrict__ Kb,
                                 __bf16* __restrict__ VTb,
                                 int Bn, int T, int C, int H, int D) {
    const size_t total = (size_t)Bn * T * (C / 2);
    size_t idx = (size_t)blockIdx.x * blockDim.x + threadIdx.x;
    if (idx >= total) return;
    const int c2 = (int)(idx % (size_t)(C / 2));
    const size_t bt = idx / (size_t)(C / 2);
    const int t = (int)(bt % (size_t)T);
    const int b = (int)(bt / (size_t)T);

    const float inv_freq = __expf(-((float)(2 * c2) / (float)C) * 9.210340372f);
    const float ang = (float)t * inv_freq;
    float sn, cs;
    __sincosf(ang, &sn, &cs);

    const size_t src = bt * (size_t)C + (size_t)(2 * c2);
    const int c = 2 * c2;
    const int h = c / D;
    const int d = c % D;
    const size_t qdst = (((size_t)b * H + h) * (size_t)T + t) * (size_t)D + d;

    float xe = qf[src], xo = qf[src + 1];
    Qb[qdst]     = (__bf16)(xe * cs - xo * sn);
    Qb[qdst + 1] = (__bf16)(xe * sn + xo * cs);

    xe = kf[src]; xo = kf[src + 1];
    Kb[qdst]     = (__bf16)(xe * cs - xo * sn);
    Kb[qdst + 1] = (__bf16)(xe * sn + xo * cs);

    const size_t vdst = (((size_t)b * H + h) * (size_t)D + d) * (size_t)T + t;
    VTb[vdst]     = (__bf16)vf[src];
    VTb[vdst + T] = (__bf16)vf[src + 1];
}

// ---------------------------------------------------------------------------
// Causal flash attention with block-shared K/V tiles staged via async->LDS.
// Grid: (T/128, H, B). Block = 256 thr = 8 waves; wave owns 16 query rows.
// All waves run the same tile loop (uniform barriers); masked waves skip math.
// ---------------------------------------------------------------------------
__global__ __launch_bounds__(256) void flash_attn_kernel(
    const __bf16* __restrict__ Qb, const __bf16* __restrict__ Kb,
    const __bf16* __restrict__ VTb, __bf16* __restrict__ Ob,
    int T, int H, int D, int C) {
    __shared__ __align__(16) __bf16 ldsK[64 * 64];   // [key][d]    8 KB
    __shared__ __align__(16) __bf16 ldsV[64 * 64];   // [d][key]    8 KB
    __shared__ __align__(16) __bf16 ldsP[8][16 * 64];//            16 KB

    const int tid  = threadIdx.x;
    const int lane = tid & 31;
    const int wave = tid >> 5;
    const int half = lane >> 4;
    const int r    = lane & 15;
    const int b = blockIdx.z, h = blockIdx.y;
    const int q0 = blockIdx.x * 128 + wave * 16;
    const size_t bhT = ((size_t)b * H + h) * (size_t)T;
    const size_t bhD = ((size_t)b * H + h) * (size_t)D;

    // Q fragments (rows q0..q0+15, D=64 -> two K=32 steps), loaded once.
    const __bf16* qrow = Qb + (bhT + q0 + r) * (size_t)D;
    const bf16x16 qf0 = load_frag_a(qrow, 0, half);
    const bf16x16 qf1 = load_frag_a(qrow, 32, half);

    f32x8 o[4] = {};
    float mrow[8], lrow[8];
#pragma unroll
    for (int j = 0; j < 8; ++j) { mrow[j] = -3.0e38f; lrow[j] = 0.0f; }

    const float scale = 0.125f;              // 1/sqrt(64)
    const int kb_top = 2 * blockIdx.x + 1;   // block-uniform causal extent

    for (int kb = 0; kb <= kb_top; ++kb) {
        const int kbase = kb * 64;

        __syncthreads();  // previous tile fully consumed
        // Cooperative stage of K tile [64 keys][64 d] and V tile [64 d][64 keys]:
        // 8 KB each = 512 16-byte chunks; 256 threads x 2 chunks.
#pragma unroll
        for (int c = tid; c < 512; c += 256) {
            const int row  = c >> 3;
            const int col8 = (c & 7) * 8;
            copy16_to_lds(Kb  + (bhT + kbase + row) * 64 + col8, ldsK + row * 64 + col8);
            copy16_to_lds(VTb + (bhD + row) * (size_t)T + kbase + col8,
                          ldsV + row * 64 + col8);
        }
        wait_async_copies();
        __syncthreads();  // tile visible to all waves

        if (kbase > q0 + 15) continue;  // this wave's rows are fully masked

        // S = Q K^T for 64 keys: B-frags from LDS (ds_load_b128)
        f32x8 s[4];
#pragma unroll
        for (int nt = 0; nt < 4; ++nt) {
            const __bf16* krow = ldsK + (nt * 16 + r) * 64;
            bf16x16 kf0 = load_frag_b(krow, 0, half);
            bf16x16 kf1 = load_frag_b(krow, 32, half);
            f32x8 z = {};
            z = WMMA_BF16(qf0, kf0, z);
            s[nt] = WMMA_BF16(qf1, kf1, z);
        }

        // scale + causal mask
        const bool needmask = (kbase + 63) > q0;
#pragma unroll
        for (int nt = 0; nt < 4; ++nt)
#pragma unroll
            for (int j = 0; j < 8; ++j) {
                float v = s[nt][j] * scale;
                if (needmask) {
                    const int m = q0 + j + 8 * half;
                    const int n = kbase + nt * 16 + r;
                    if (n > m) v = -1.0e30f;
                }
                s[nt][j] = v;
            }

        // online softmax: row max across n-tiles + 16-lane half reduction
        float rm[8];
#pragma unroll
        for (int j = 0; j < 8; ++j)
            rm[j] = fmaxf(fmaxf(s[0][j], s[1][j]), fmaxf(s[2][j], s[3][j]));
#pragma unroll
        for (int msk = 1; msk < 16; msk <<= 1)
#pragma unroll
            for (int j = 0; j < 8; ++j)
                rm[j] = fmaxf(rm[j], __shfl_xor(rm[j], msk, 32));

        float mnew[8], alpha[8], rsum[8];
#pragma unroll
        for (int j = 0; j < 8; ++j) {
            mnew[j]  = fmaxf(mrow[j], rm[j]);
            alpha[j] = __expf(mrow[j] - mnew[j]);
            rsum[j]  = 0.0f;
        }
#pragma unroll
        for (int nt = 0; nt < 4; ++nt)
#pragma unroll
            for (int j = 0; j < 8; ++j) {
                const float p = __expf(s[nt][j] - mnew[j]);
                s[nt][j] = p;
                rsum[j] += p;
            }
#pragma unroll
        for (int msk = 1; msk < 16; msk <<= 1)
#pragma unroll
            for (int j = 0; j < 8; ++j)
                rsum[j] += __shfl_xor(rsum[j], msk, 32);
#pragma unroll
        for (int j = 0; j < 8; ++j) {
            lrow[j] = lrow[j] * alpha[j] + rsum[j];
            mrow[j] = mnew[j];
        }
#pragma unroll
        for (int dt = 0; dt < 4; ++dt)
#pragma unroll
            for (int j = 0; j < 8; ++j) o[dt][j] *= alpha[j];

        // P: C-layout -> A-layout via per-wave LDS slab (same-wave DS in order)
        __bf16* pw = ldsP[wave];
#pragma unroll
        for (int nt = 0; nt < 4; ++nt)
#pragma unroll
            for (int j = 0; j < 8; ++j)
                pw[(j + 8 * half) * 64 + nt * 16 + r] = (__bf16)s[nt][j];
        asm volatile("s_wait_dscnt 0" ::: "memory");

        const __bf16* prow = pw + r * 64;
        bf16x16 pf0 = load_frag_a(prow, 0, half);
        bf16x16 pf1 = load_frag_a(prow, 32, half);

#pragma unroll
        for (int dt = 0; dt < 4; ++dt) {
            const __bf16* vrow = ldsV + (dt * 16 + r) * 64;
            bf16x16 vf0 = load_frag_b(vrow, 0, half);
            bf16x16 vf1 = load_frag_b(vrow, 32, half);
            o[dt] = WMMA_BF16(pf0, vf0, o[dt]);
            o[dt] = WMMA_BF16(pf1, vf1, o[dt]);
        }
    }

    // epilogue: normalize, write [B,T,H*D] bf16
#pragma unroll
    for (int dt = 0; dt < 4; ++dt)
#pragma unroll
        for (int j = 0; j < 8; ++j) {
            const int m = q0 + j + 8 * half;
            Ob[((size_t)b * T + m) * (size_t)C + h * 64 + dt * 16 + r] =
                (__bf16)(o[dt][j] / lrow[j]);
        }
}

// ---------------------------------------------------------------------------
// Host orchestration
// ---------------------------------------------------------------------------
extern "C" void kernel_launch(void* const* d_in, const int* in_sizes, int n_in,
                              void* d_out, int out_size, void* d_ws, size_t ws_size,
                              hipStream_t stream) {
    constexpr int Bn = 4, T = 2048, C = 512, H = 8, D = 64;
    constexpr size_t nX = (size_t)Bn * T * C;
    constexpr size_t nW = (size_t)C * C;
    constexpr int M = Bn * T;

    const float* x  = (const float*)d_in[0];
    const float* Wq = (const float*)d_in[1];
    const float* Wk = (const float*)d_in[2];
    const float* Wv = (const float*)d_in[3];
    const float* Wp = (const float*)d_in[4];
    const float* bp = (const float*)d_in[5];
    float* out = (float*)d_out;

    char* p = (char*)d_ws;
    auto take = [&](size_t bytes) { char* q = p; p += (bytes + 255) & ~(size_t)255; return q; };
    __bf16* xb  = (__bf16*)take(nX * 2);
    __bf16* Wqb = (__bf16*)take(nW * 2);
    __bf16* Wkb = (__bf16*)take(nW * 2);
    __bf16* Wvb = (__bf16*)take(nW * 2);
    __bf16* Wpb = (__bf16*)take(nW * 2);
    float*  qf  = (float*)take(nX * 4);
    float*  kf  = (float*)take(nX * 4);
    float*  vf  = (float*)take(nX * 4);
    __bf16* Qb  = (__bf16*)take(nX * 2);
    __bf16* Kb  = (__bf16*)take(nX * 2);
    __bf16* VTb = (__bf16*)take(nX * 2);
    __bf16* Ab  = (__bf16*)take(nX * 2);
    (void)ws_size; (void)in_sizes; (void)n_in; (void)out_size;

    {
        const int thr = 256;
        f32_to_bf16_kernel<<<(unsigned)((nX / 4 + thr - 1) / thr), thr, 0, stream>>>(x,  xb,  nX / 4);
        f32_to_bf16_kernel<<<(unsigned)((nW / 4 + thr - 1) / thr), thr, 0, stream>>>(Wq, Wqb, nW / 4);
        f32_to_bf16_kernel<<<(unsigned)((nW / 4 + thr - 1) / thr), thr, 0, stream>>>(Wk, Wkb, nW / 4);
        f32_to_bf16_kernel<<<(unsigned)((nW / 4 + thr - 1) / thr), thr, 0, stream>>>(Wv, Wvb, nW / 4);
        f32_to_bf16_kernel<<<(unsigned)((nW / 4 + thr - 1) / thr), thr, 0, stream>>>(Wp, Wpb, nW / 4);
    }

    {
        dim3 grid(C / 64, M / 128);
        gemm_bf16_nt_kernel<<<grid, 256, 0, stream>>>(xb, Wqb, nullptr, qf, M, C, C);
        gemm_bf16_nt_kernel<<<grid, 256, 0, stream>>>(xb, Wkb, nullptr, kf, M, C, C);
        gemm_bf16_nt_kernel<<<grid, 256, 0, stream>>>(xb, Wvb, nullptr, vf, M, C, C);
    }

    {
        const size_t total = (size_t)Bn * T * (C / 2);
        rope_pack_kernel<<<(unsigned)((total + 255) / 256), 256, 0, stream>>>(
            qf, kf, vf, Qb, Kb, VTb, Bn, T, C, H, D);
    }

    {
        dim3 grid(T / 128, H, Bn);
        flash_attn_kernel<<<grid, 256, 0, stream>>>(Qb, Kb, VTb, Ab, T, H, D, C);
    }

    {
        dim3 grid(C / 64, M / 128);
        gemm_bf16_nt_kernel<<<grid, 256, 0, stream>>>(Ab, Wpb, bp, out, M, C, C);
    }
}
